// Hawk_3547642987361
// MI455X (gfx1250) — matrix operand order repaired and checked by reading, not compiled, not run on hardware
//
#include <hip/hip_runtime.h>
#include <hip/hip_bf16.h>
#include <math.h>
#include <stdint.h>

// ---------------- problem constants ----------------
#define BDIM   4
#define TLEN   4096
#define DIM    1024
#define HID    1536
#define N1     (2 * HID)       // 3072
#define KCONV  4
#define BT     (BDIM * TLEN)   // 16384
#define CCONST 8.0f

// scan chunking
#define CH     128             // steps per chunk
#define NCH    (TLEN / CH)     // 32

// ---------------- WMMA types ----------------
typedef __attribute__((ext_vector_type(16))) __bf16 v16bf;
typedef __attribute__((ext_vector_type(8)))  float  v8f;

// ---------------- GEMM config ----------------
#define BM   128
#define BN   256
#define BK   32
#define LDA  40                       // padded LDS row stride (bf16 elements)
#define BUFE ((BM + BN) * LDA)        // elements per double-buffer stage

#if __has_builtin(__builtin_amdgcn_s_wait_asynccnt)
#define WAIT_ASYNC() __builtin_amdgcn_s_wait_asynccnt(0)
#else
#define WAIT_ASYNC() asm volatile("s_wait_asynccnt 0x0" ::: "memory")
#endif

// Y[M x N] = X[M x K] * W[N x K]^T (+ bias[N]), fp32 accum via bf16 WMMA.
// 8 waves; block tile 128x256; wave tile 64x64 (4x4 WMMA frags);
// double-buffered LDS filled with GLOBAL_LOAD_ASYNC_TO_LDS_B128.
__global__ __launch_bounds__(256) void gemm_xwT_bf16(
    const __bf16* __restrict__ X,
    const __bf16* __restrict__ W,
    const float*  __restrict__ bias,   // may be nullptr
    float* __restrict__ Y,
    int M, int N, int K)
{
    __shared__ __bf16 smem[2 * BUFE];

    const int tid  = threadIdx.x;
    const int lane = tid & 31;
    const int wid  = tid >> 5;           // 0..7
    const int wm   = (wid >> 2) * 64;    // 2 waves in M
    const int wn   = (wid & 3) * 64;     // 4 waves in N
    const int bm   = blockIdx.y * BM;
    const int bn   = blockIdx.x * BN;

    // wave-relative LDS byte base of smem (shared aperture lives in ADDR[63:32])
    const uint32_t ldsBase = (uint32_t)(uintptr_t)(&smem[0]);

    // issue one stage of async global->LDS copies (6 x b128 per thread)
    auto stage_async = [&](int buf, int kk) {
        const uint32_t aBase = ldsBase + (uint32_t)(buf * BUFE * 2);
        const uint32_t wBase = aBase + (uint32_t)(BM * LDA * 2);
#pragma unroll
        for (int i = 0; i < 2; ++i) {              // A tile: 128x32 bf16
            int idx = (tid + i * 256) * 8;
            int r = idx >> 5, c = idx & 31;
            uint32_t lds = aBase + (uint32_t)((r * LDA + c) * 2);
            const __bf16* g = X + (size_t)(bm + r) * K + kk + c;
            asm volatile("global_load_async_to_lds_b128 %0, %1, off"
                         :: "v"(lds), "v"(g) : "memory");
        }
#pragma unroll
        for (int i = 0; i < 4; ++i) {              // W tile: 256x32 bf16
            int idx = (tid + i * 256) * 8;
            int r = idx >> 5, c = idx & 31;
            uint32_t lds = wBase + (uint32_t)((r * LDA + c) * 2);
            const __bf16* g = W + (size_t)(bn + r) * K + kk + c;
            asm volatile("global_load_async_to_lds_b128 %0, %1, off"
                         :: "v"(lds), "v"(g) : "memory");
        }
    };

    v8f acc[4][4];
#pragma unroll
    for (int i = 0; i < 4; ++i)
#pragma unroll
        for (int j = 0; j < 4; ++j) acc[i][j] = {};

    const int arow = lane & 15;
    const int kbA  = (lane >> 4) * 8;    // A frag: lanes 0-15 -> K {0..7,16..23}; 16-31 -> {8..15,24..31}
    const int kbB  = (lane >> 4) * 8;    // B frag: lanes 0-15 -> K {0..7,16..23}; 16-31 -> {8..15,24..31}

    stage_async(0, 0);                   // prologue

    int buf = 0;
    for (int kk = 0; kk < K; kk += BK, buf ^= 1) {
        WAIT_ASYNC();                    // my copies for buf done
        __syncthreads();                 // everyone's copies done; prev reads of other buf done
        if (kk + BK < K) stage_async(buf ^ 1, kk + BK);   // overlap next stage with compute

        const __bf16* Abase = &smem[buf * BUFE];
        const __bf16* Wbase = Abase + BM * LDA;

        v16bf afrag[4];
#pragma unroll
        for (int i = 0; i < 4; ++i) {
            const __bf16* p = &Abase[(wm + i * 16 + arow) * LDA];
            v16bf f;
#pragma unroll
            for (int q = 0; q < 4; ++q) {
                f[2 * q]         = p[kbA + 2 * q];
                f[2 * q + 1]     = p[kbA + 2 * q + 1];
                f[8 + 2 * q]     = p[kbA + 16 + 2 * q];
                f[8 + 2 * q + 1] = p[kbA + 16 + 2 * q + 1];
            }
            afrag[i] = f;
        }
        v16bf bfrag[4];
#pragma unroll
        for (int j = 0; j < 4; ++j) {
            const __bf16* p = &Wbase[(wn + j * 16 + arow) * LDA];
            v16bf f;
#pragma unroll
            for (int q = 0; q < 16; ++q) f[q] = p[kbB + (q & 7) + ((q >> 3) << 4)];
            bfrag[j] = f;
        }

#pragma unroll
        for (int i = 0; i < 4; ++i)
#pragma unroll
            for (int j = 0; j < 4; ++j)
                acc[i][j] = __builtin_amdgcn_wmma_f32_16x16x32_bf16(
                    false, afrag[i], false, bfrag[j], (short)0, acc[i][j], false, false);
    }

    // C/D layout: N = lane&15 ; M = r + 8*(lane>>4)
    const int mofs = (lane >> 4) * 8;
    const int ccol = lane & 15;
#pragma unroll
    for (int i = 0; i < 4; ++i)
#pragma unroll
        for (int j = 0; j < 4; ++j) {
            int gn = bn + wn + j * 16 + ccol;
            float bv = bias ? bias[gn] : 0.0f;
#pragma unroll
            for (int r = 0; r < 8; ++r) {
                int gm = bm + wm + i * 16 + mofs + r;
                Y[(size_t)gm * N + gn] = acc[i][j][r] + bv;
            }
        }
}

// ---------------- fp32 -> bf16 conversion ----------------
__global__ void f32_to_bf16(const float* __restrict__ in, __bf16* __restrict__ out, size_t n)
{
    size_t i = (size_t)blockIdx.x * blockDim.x + threadIdx.x;
    if (i < n) out[i] = (__bf16)in[i];
}

// ---------------- depthwise causal conv (K=4) ----------------
// xc[b,t,h] = sum_k xh[b,t-3+k,h]*w[h,k] + cb[h]; xh = gx[..., HID: ]
__global__ void conv_kernel(const float* __restrict__ gx,
                            const float* __restrict__ cw,
                            const float* __restrict__ cb,
                            float* __restrict__ xc,
                            __bf16* __restrict__ xc_bf)
{
    size_t idx = (size_t)blockIdx.x * blockDim.x + threadIdx.x;
    if (idx >= (size_t)BT * HID) return;
    int h  = (int)(idx % HID);
    int bt = (int)(idx / HID);
    int t  = bt % TLEN;
    int b  = bt / TLEN;
    float acc = cb[h];
#pragma unroll
    for (int k = 0; k < KCONV; ++k) {
        int tt = t - (KCONV - 1) + k;
        if (tt >= 0)
            acc += gx[((size_t)(b * TLEN + tt)) * N1 + HID + h] * cw[h * KCONV + k];
    }
    xc[idx]    = acc;
    xc_bf[idx] = (__bf16)acc;
}

// ---------------- pointwise: alpha / xbeta (in-place over logits) ----------------
__global__ void gates_kernel(float* __restrict__ recl,   // in: gp logits, out: alpha
                             float* __restrict__ inpl,   // in: ip logits, out: xbeta
                             const float* __restrict__ xc,
                             const float* __restrict__ flam)
{
    size_t idx = (size_t)blockIdx.x * blockDim.x + threadIdx.x;
    if (idx >= (size_t)BT * HID) return;
    int h = (int)(idx % HID);
    float rec = 1.0f / (1.0f + expf(-recl[idx]));
    float inp = 1.0f / (1.0f + expf(-inpl[idx]));
    float sp  = log1pf(expf(flam[h]));          // softplus
    float alpha = expf(-CCONST * sp * rec);
    float beta  = sqrtf(1.0f - alpha * alpha + 1e-6f);
    recl[idx] = alpha;
    inpl[idx] = beta * inp * xc[idx];
}

// ---------------- 3-phase chunked linear scan ----------------
__global__ void scan_phase1(const float* __restrict__ alpha,
                            const float* __restrict__ xbeta,
                            float* __restrict__ chunkA,
                            float* __restrict__ chunkB)
{
    size_t idx = (size_t)blockIdx.x * blockDim.x + threadIdx.x;
    if (idx >= (size_t)BDIM * NCH * HID) return;
    int h = (int)(idx % HID);
    int c = (int)((idx / HID) % NCH);
    int b = (int)(idx / ((size_t)HID * NCH));
    size_t base = ((size_t)b * TLEN + (size_t)c * CH) * HID + h;
    float A = 1.0f, Bv = 0.0f;
    for (int s = 0; s < CH; ++s) {
        float a = alpha[base + (size_t)s * HID];
        float x = xbeta[base + (size_t)s * HID];
        Bv = a * Bv + x;
        A *= a;
    }
    chunkA[idx] = A;
    chunkB[idx] = Bv;
}

__global__ void scan_phase2(const float* __restrict__ chunkA,
                            const float* __restrict__ chunkB,
                            float* __restrict__ carry)
{
    size_t idx = (size_t)blockIdx.x * blockDim.x + threadIdx.x;
    if (idx >= (size_t)BDIM * HID) return;
    int h = (int)(idx % HID);
    int b = (int)(idx / HID);
    float run = 0.0f;
    for (int c = 0; c < NCH; ++c) {
        size_t o = ((size_t)b * NCH + c) * HID + h;
        carry[o] = run;
        run = chunkA[o] * run + chunkB[o];
    }
}

__global__ void scan_phase3(const float* __restrict__ alpha,
                            const float* __restrict__ xbeta,
                            const float* __restrict__ carry,
                            const float* __restrict__ gx,   // gate = gx[..., :HID]
                            __bf16* __restrict__ G)
{
    size_t idx = (size_t)blockIdx.x * blockDim.x + threadIdx.x;
    if (idx >= (size_t)BDIM * NCH * HID) return;
    int h = (int)(idx % HID);
    int c = (int)((idx / HID) % NCH);
    int b = (int)(idx / ((size_t)HID * NCH));
    size_t base = ((size_t)b * TLEN + (size_t)c * CH) * HID + h;
    float hv = carry[((size_t)b * NCH + c) * HID + h];
    for (int s = 0; s < CH; ++s) {
        int bt = b * TLEN + c * CH + s;
        hv = alpha[base + (size_t)s * HID] * hv + xbeta[base + (size_t)s * HID];
        float g  = gx[(size_t)bt * N1 + h];
        float gl = 0.5f * g * (1.0f + erff(g * 0.70710678f));   // exact gelu
        G[(size_t)bt * HID + h] = (__bf16)(gl * hv);
    }
}

// ---------------- host side ----------------
static inline size_t align_up(size_t x) { return (x + 255) & ~(size_t)255; }

extern "C" void kernel_launch(void* const* d_in, const int* in_sizes, int n_in,
                              void* d_out, int out_size, void* d_ws, size_t ws_size,
                              hipStream_t stream)
{
    (void)in_sizes; (void)n_in; (void)out_size; (void)ws_size;
    const float* x      = (const float*)d_in[0];
    const float* proj_w = (const float*)d_in[1];
    const float* conv_w = (const float*)d_in[2];
    const float* conv_b = (const float*)d_in[3];
    const float* ip_w   = (const float*)d_in[4];
    const float* ip_b   = (const float*)d_in[5];
    const float* gp_w   = (const float*)d_in[6];
    const float* gp_b   = (const float*)d_in[7];
    const float* flam   = (const float*)d_in[8];
    const float* out_w  = (const float*)d_in[9];
    float* out = (float*)d_out;

    char* p = (char*)d_ws;
    size_t off = 0;
    auto grab = [&](size_t bytes) { void* r = p + off; off += align_up(bytes); return r; };

    __bf16* x_bf     = (__bf16*)grab((size_t)BT * DIM * 2);
    __bf16* wproj_bf = (__bf16*)grab((size_t)N1 * DIM * 2);
    __bf16* wgp_bf   = (__bf16*)grab((size_t)HID * HID * 2);
    __bf16* wip_bf   = (__bf16*)grab((size_t)HID * HID * 2);
    __bf16* wout_bf  = (__bf16*)grab((size_t)DIM * HID * 2);
    float*  gx       = (float*) grab((size_t)BT * N1 * 4);
    float*  xc       = (float*) grab((size_t)BT * HID * 4);
    __bf16* xc_bf    = (__bf16*)grab((size_t)BT * HID * 2);
    float*  recl     = (float*) grab((size_t)BT * HID * 4);   // -> alpha
    float*  inpl     = (float*) grab((size_t)BT * HID * 4);   // -> xbeta
    float*  chunkA   = (float*) grab((size_t)BDIM * NCH * HID * 4);
    float*  chunkB   = (float*) grab((size_t)BDIM * NCH * HID * 4);
    float*  carry    = (float*) grab((size_t)BDIM * NCH * HID * 4);
    __bf16* G_bf     = (__bf16*)grab((size_t)BT * HID * 2);

    const int TPB = 256;
    auto blocks = [](size_t n, int t) { return (unsigned)((n + t - 1) / t); };

    // 1) fp32 -> bf16 conversions
    f32_to_bf16<<<blocks((size_t)BT * DIM, TPB), TPB, 0, stream>>>(x, x_bf, (size_t)BT * DIM);
    f32_to_bf16<<<blocks((size_t)N1 * DIM, TPB), TPB, 0, stream>>>(proj_w, wproj_bf, (size_t)N1 * DIM);
    f32_to_bf16<<<blocks((size_t)HID * HID, TPB), TPB, 0, stream>>>(gp_w, wgp_bf, (size_t)HID * HID);
    f32_to_bf16<<<blocks((size_t)HID * HID, TPB), TPB, 0, stream>>>(ip_w, wip_bf, (size_t)HID * HID);
    f32_to_bf16<<<blocks((size_t)DIM * HID, TPB), TPB, 0, stream>>>(out_w, wout_bf, (size_t)DIM * HID);

    // 2) projection GEMM: gx = x @ proj_w^T   (16384 x 3072, K=1024)
    gemm_xwT_bf16<<<dim3(N1 / BN, BT / BM), TPB, 0, stream>>>(
        x_bf, wproj_bf, nullptr, gx, BT, N1, DIM);

    // 3) depthwise causal conv + bias
    conv_kernel<<<blocks((size_t)BT * HID, TPB), TPB, 0, stream>>>(gx, conv_w, conv_b, xc, xc_bf);

    // 4) gate GEMMs (16384 x 1536, K=1536)
    gemm_xwT_bf16<<<dim3(HID / BN, BT / BM), TPB, 0, stream>>>(
        xc_bf, wgp_bf, gp_b, recl, BT, HID, HID);
    gemm_xwT_bf16<<<dim3(HID / BN, BT / BM), TPB, 0, stream>>>(
        xc_bf, wip_bf, ip_b, inpl, BT, HID, HID);

    // 5) pointwise alpha/beta/xbeta
    gates_kernel<<<blocks((size_t)BT * HID, TPB), TPB, 0, stream>>>(recl, inpl, xc, flam);

    // 6) chunked linear scan + fused gelu(gate)*h -> bf16
    scan_phase1<<<blocks((size_t)BDIM * NCH * HID, TPB), TPB, 0, stream>>>(recl, inpl, chunkA, chunkB);
    scan_phase2<<<blocks((size_t)BDIM * HID, TPB), TPB, 0, stream>>>(chunkA, chunkB, carry);
    scan_phase3<<<blocks((size_t)BDIM * NCH * HID, TPB), TPB, 0, stream>>>(recl, inpl, carry, gx, G_bf);

    // 7) output GEMM: out = G @ out_w^T   (16384 x 1024, K=1536)
    gemm_xwT_bf16<<<dim3(DIM / BN, BT / BM), TPB, 0, stream>>>(
        G_bf, wout_bf, nullptr, out, BT, DIM, HID);
}
